// BatchedLinearHolomorphicEQProp_37606733643852
// MI455X (gfx1250) — compile-verified, matrix operand order checked by reading, not run on hardware
//
#include <hip/hip_runtime.h>
#include <math.h>

#define IN_SZ   512
#define HID_SZ  8192
#define OUT_SZ  512
#define TOTAL   9216
#define BATCH   16

#define NT_TILES 544          // (TOTAL - IN_SZ) / 16 : only non-clamped columns need g
#define KT_TILES 288          // TOTAL / 32
#define CHUNKS   8
#define KT_PER_CHUNK 36       // 288 / 8 (even -> clean dual-accumulator split)
#define K_PER_CHUNK  1152     // 36 * 32
#define LDS_STRIDE   1160     // 1152 + 8 bf16 pad -> row stride 2320 B, kills bank conflicts

#define WAVES_PER_BLOCK 4
#define BLOCK_THREADS   128
#define GRID_BLOCKS     (NT_TILES / WAVES_PER_BLOCK)   // 136

typedef __attribute__((ext_vector_type(16))) __bf16 v16bf;
typedef __attribute__((ext_vector_type(8)))  __bf16 v8bf;
typedef __attribute__((ext_vector_type(8)))  float  v8f;

__device__ __forceinline__ float fast_tanh(float x) {
#if __has_builtin(__builtin_amdgcn_tanhf)
  return __builtin_amdgcn_tanhf(x);          // single v_tanh_f32
#elif __has_builtin(__builtin_amdgcn_tanh_f32)
  return __builtin_amdgcn_tanh_f32(x);
#else
  return tanhf(x);
#endif
}

// ---------------------------------------------------------------------------
// Pack Wsym = 0.5*(W + W^T) into bf16 WMMA B-fragment tiles.
// Tile (nt, kt) covers columns n = IN_SZ + nt*16 + (lane&15),
// rows k = kt*32 + (i<8?0:16) + (lane>>4)*8 + (i&7)  (16x16x32 bf16 B layout).
// Each lane stores its 16 bf16 (32 bytes) contiguously -> streaming b128 loads.
// ---------------------------------------------------------------------------
__global__ __launch_bounds__(256) void pack_wsym(const float* __restrict__ W,
                                                 __bf16* __restrict__ P) {
  long long t = (long long)blockIdx.x * 256 + threadIdx.x;
  long long tile = t >> 5;
  if (tile >= (long long)NT_TILES * KT_TILES) return;
  int lane = (int)(t & 31);
  int kt = (int)(tile % KT_TILES);
  int nt = (int)(tile / KT_TILES);
  int n = IN_SZ + nt * 16 + (lane & 15);
  int kbase = kt * 32 + (lane >> 4) * 8;
  __bf16* dst = P + (tile * 32LL + lane) * 16;
#pragma unroll
  for (int i = 0; i < 16; ++i) {
    int k = kbase + ((i < 8) ? 0 : 16) + (i & 7);
    float v = 0.5f * (W[(size_t)k * TOTAL + n] + W[(size_t)n * TOTAL + k]);
    dst[i] = (__bf16)v;
  }
}

// ---------------------------------------------------------------------------
// Init both ping-pong state buffers (fp32 master + bf16 mirror).
// ---------------------------------------------------------------------------
__global__ __launch_bounds__(256) void init_s(const float* __restrict__ x,
                                              float* s0f, float* s1f,
                                              __bf16* s0b, __bf16* s1b) {
  int t = blockIdx.x * 256 + threadIdx.x;
  if (t >= BATCH * TOTAL) return;
  int m = t / TOTAL, c = t % TOTAL;
  float v = (c < IN_SZ) ? x[m * IN_SZ + c] : 0.0f;
  __bf16 h = (__bf16)v;
  s0f[t] = v; s1f[t] = v; s0b[t] = h; s1b[t] = h;
}

// ---------------------------------------------------------------------------
// Device-wide sense barrier (all 136 blocks co-resident).
// ---------------------------------------------------------------------------
__device__ __forceinline__ void grid_barrier(volatile unsigned* cnt,
                                             volatile unsigned* gen,
                                             unsigned nblocks) {
  __syncthreads();
  if (threadIdx.x == 0) {
    unsigned g = *gen;
    __threadfence();
    unsigned prev = atomicAdd((unsigned*)cnt, 1u);
    if (prev == nblocks - 1u) {
      *cnt = 0u;
      __threadfence();
      atomicAdd((unsigned*)gen, 1u);
    } else {
      while (*gen == g) { __builtin_amdgcn_s_sleep(1); }
    }
    __threadfence();
  }
  __syncthreads();
}

// ---------------------------------------------------------------------------
// One settle iteration. Called with *fixed* pointer direction so every memory
// op is provably global address-space (no flat_load in the update tail).
// ---------------------------------------------------------------------------
__device__ __forceinline__ void settle_iter(
    const __bf16* __restrict__ P,
    const float* __restrict__ sin_f, const __bf16* __restrict__ sin_b,
    float* __restrict__ sout_f, __bf16* __restrict__ sout_b,
    float* __restrict__ out, bool last,
    __bf16* sA, int tid, int lane, int nt, int half, int nloc, int n) {
  v8f acc0 = {}, acc1 = {};
  for (int ch = 0; ch < CHUNKS; ++ch) {
    const int c0 = ch * K_PER_CHUNK;
    __syncthreads();   // previous chunk fully consumed before restaging
    for (int j = tid; j < BATCH * (K_PER_CHUNK / 8); j += BLOCK_THREADS) {
      int row = j / (K_PER_CHUNK / 8);
      int g8  = j % (K_PER_CHUNK / 8);
      *(uint4*)(sA + row * LDS_STRIDE + g8 * 8) =
          *(const uint4*)(sin_b + (size_t)row * TOTAL + c0 + g8 * 8);
    }
    __syncthreads();

    const __bf16* arow = sA + nloc * LDS_STRIDE + half * 8;
    const __bf16* bp   = P + ((size_t)nt * KT_TILES + ch * KT_PER_CHUNK) * 512
                           + lane * 16;
#pragma unroll 2
    for (int ktl = 0; ktl < KT_PER_CHUNK; ktl += 2) {
      union { v16bf v; v8bf h[2]; } a0, a1;
      a0.h[0] = *(const v8bf*)(arow + ktl * 32);
      a0.h[1] = *(const v8bf*)(arow + ktl * 32 + 16);
      a1.h[0] = *(const v8bf*)(arow + ktl * 32 + 32);
      a1.h[1] = *(const v8bf*)(arow + ktl * 32 + 48);
      v16bf b0 = *(const v16bf*)bp;
      v16bf b1 = *(const v16bf*)(bp + 512);
      bp += 1024;
      __builtin_prefetch(bp, 0, 1);        // global_prefetch_b8, next B tiles
      acc0 = __builtin_amdgcn_wmma_f32_16x16x32_bf16(
          false, a0.v, false, b0, (short)0, acc0, false, false);
      acc1 = __builtin_amdgcn_wmma_f32_16x16x32_bf16(
          false, a1.v, false, b1, (short)0, acc1, false, false);
    }
  }

#pragma unroll
  for (int r = 0; r < 8; ++r) {
    const int m = half * 8 + r;
    float g  = acc0[r] + acc1[r];
    float sv = sin_f[(size_t)m * TOTAL + n];
    float ns = fast_tanh(sv - 0.5f * g);
    sout_f[(size_t)m * TOTAL + n] = ns;
    sout_b[(size_t)m * TOTAL + n] = (__bf16)ns;
    if (last && n >= IN_SZ + HID_SZ) {
      out[(size_t)m * OUT_SZ + (n - (IN_SZ + HID_SZ))] = ns;
    }
  }
}

// ---------------------------------------------------------------------------
// Persistent settle kernel: T iterations of s <- tanh(s - 0.5 * s@Wsym) on the
// hidden+output slices, device-wide barrier between iterations. Time loop is
// two-phase so each phase has compile-time-fixed ping/pong pointers.
// ---------------------------------------------------------------------------
__global__ __launch_bounds__(BLOCK_THREADS) void settle(
    const __bf16* __restrict__ P,
    float* __restrict__ s0f, __bf16* __restrict__ s0b,
    float* __restrict__ s1f, __bf16* __restrict__ s1b,
    const int* __restrict__ Tp,
    unsigned* bar_cnt, unsigned* bar_gen,
    float* __restrict__ out) {
  __shared__ __align__(16) __bf16 sA[BATCH * LDS_STRIDE];   // 37120 B

  const int tid  = threadIdx.x;
  const int lane = tid & 31;
  const int wave = tid >> 5;
  const int nt   = blockIdx.x * WAVES_PER_BLOCK + wave;     // 0..543
  const int half = lane >> 4;
  const int nloc = lane & 15;
  const int n    = IN_SZ + nt * 16 + nloc;

  const int T = *Tp;

  int t = 0;
  while (t < T) {
    settle_iter(P, s0f, s0b, s1f, s1b, out, t == T - 1,
                sA, tid, lane, nt, half, nloc, n);
    grid_barrier(bar_cnt, bar_gen, gridDim.x);
    ++t;
    if (t >= T) break;
    settle_iter(P, s1f, s1b, s0f, s0b, out, t == T - 1,
                sA, tid, lane, nt, half, nloc, n);
    grid_barrier(bar_cnt, bar_gen, gridDim.x);
    ++t;
  }
}

// ---------------------------------------------------------------------------
extern "C" void kernel_launch(void* const* d_in, const int* in_sizes, int n_in,
                              void* d_out, int out_size, void* d_ws, size_t ws_size,
                              hipStream_t stream) {
  (void)in_sizes; (void)n_in; (void)out_size; (void)ws_size;
  const float* x = (const float*)d_in[0];
  const float* W = (const float*)d_in[1];
  const int* Tp  = (const int*)d_in[2];   // T_settle, read on-device

  char* ws = (char*)d_ws;
  size_t o = 0;
  auto a256 = [](size_t v) { return (v + 255) & ~(size_t)255; };

  __bf16* P = (__bf16*)(ws + o);
  o = a256(o + (size_t)NT_TILES * KT_TILES * 512 * sizeof(__bf16));  // ~160.4 MB
  float* s0f = (float*)(ws + o);  o = a256(o + (size_t)BATCH * TOTAL * sizeof(float));
  float* s1f = (float*)(ws + o);  o = a256(o + (size_t)BATCH * TOTAL * sizeof(float));
  __bf16* s0b = (__bf16*)(ws + o); o = a256(o + (size_t)BATCH * TOTAL * sizeof(__bf16));
  __bf16* s1b = (__bf16*)(ws + o); o = a256(o + (size_t)BATCH * TOTAL * sizeof(__bf16));
  unsigned* bar = (unsigned*)(ws + o); o += 256;   // [0]=count, [1]=generation

  hipMemsetAsync(bar, 0, 2 * sizeof(unsigned), stream);

  {
    long long threads = (long long)NT_TILES * KT_TILES * 32;
    pack_wsym<<<(unsigned)((threads + 255) / 256), 256, 0, stream>>>(W, P);
  }
  init_s<<<(BATCH * TOTAL + 255) / 256, 256, 0, stream>>>(x, s0f, s1f, s0b, s1b);

  settle<<<GRID_BLOCKS, BLOCK_THREADS, 0, stream>>>(
      P, s0f, s0b, s1f, s1b, Tp, &bar[0], &bar[1], (float*)d_out);
}
